// DecomposeBlock_49117245997989
// MI455X (gfx1250) — matrix operand order repaired and checked by reading, not compiled, not run on hardware
//
#include <hip/hip_runtime.h>
#include <hip/hip_bf16.h>

// ---------------------------------------------------------------------------
// MI455X (gfx1250) implementation of the 12-step autoregressive rollout.
// One wave32 owns 16 rows (= 4 agent-groups) for the entire rollout.
// Decoder MLPs run on V_WMMA_F32_16X16X32_F16; GAT/TCN are scalar VALU.
// Weights are pre-packed into *fragment-order* f16 pairs so each lane loads
// its B fragment as two global_load_b128; A fragments load as 2x ds_load_b128
// and are hoisted across all N-tiles of a layer.
// ---------------------------------------------------------------------------

typedef __attribute__((ext_vector_type(16))) _Float16     v16h;
typedef __attribute__((ext_vector_type(8)))  float        v8f;
typedef __attribute__((ext_vector_type(8)))  unsigned int v8u;
typedef __attribute__((ext_vector_type(4)))  unsigned int v4u;

#define N_ROWS   40960
#define FUTURE   12
#define WAVES    2          // waves per block (64 threads)

// ---- packed-weight workspace offsets (u32 units) ----
#define OFF_Y1 0            // 84x256 -> 128x256 : 16384
#define OFF_Y2 16384        // 256x128           : 16384
#define OFF_Y3 32768        // 128x4  -> 128x16  : 1024
#define OFF_H1 33792        // 88x256 -> 128x256 (remap rows 84..87 -> 100..103)
#define OFF_H2 50176
#define OFF_H3 66560
#define OFF_S1 67584        // 104x256 -> 128x256
#define OFF_S2 83968
#define OFF_S3 100352       // 128x16 -> 128x16

// ===========================================================================
// Prep kernel: fp32 [K,N] row-major -> fragment-order packed f16 pairs.
// Fragment order: dst[ ((kc*nTiles + nt)*32 + lane)*8 + i ]  (u32 = f16 pair)
//   lane n0 = nt*16 + (lane&15); kh = (lane&16)?8:0; kp = kc*16 + kh + i
//   value = { f16(W[2kp][n0]), f16(W[2kp+1][n0]) }, zero outside [K,N].
// remapHit!=0: source row for dest-k is  k<84 -> k ; 100..103 -> 84+(k-100).
// ===========================================================================
__global__ void pack_w_kernel(const float* __restrict__ src, unsigned* __restrict__ dst,
                              int K, int N, int Kpad, int Npad, int remapHit) {
  int idx = blockIdx.x * blockDim.x + threadIdx.x;
  int total = (Kpad / 2) * Npad;
  if (idx >= total) return;
  int nTiles = Npad / 16;
  int i    = idx & 7;
  int lane = (idx >> 3) & 31;
  int tile = idx >> 8;
  int nt   = tile % nTiles;
  int kc   = tile / nTiles;
  int n    = nt * 16 + (lane & 15);
  int kh   = (lane & 16) ? 8 : 0;
  int kp   = kc * 16 + kh + i;
  int k0 = 2 * kp, k1 = 2 * kp + 1;
  int s0, s1;
  if (remapHit) {
    s0 = (k0 < 84) ? k0 : ((k0 >= 100 && k0 < 104) ? 84 + (k0 - 100) : -1);
    s1 = (k1 < 84) ? k1 : ((k1 >= 100 && k1 < 104) ? 84 + (k1 - 100) : -1);
  } else {
    s0 = (k0 < K) ? k0 : -1;
    s1 = (k1 < K) ? k1 : -1;
  }
  float v0 = 0.f, v1 = 0.f;
  if (n < N) {
    if (s0 >= 0) v0 = src[s0 * N + n];
    if (s1 >= 0) v1 = src[s1 * N + n];
  }
  union { _Float16 h[2]; unsigned u; } p;
  p.h[0] = (_Float16)v0; p.h[1] = (_Float16)v1;
  dst[idx] = p.u;
}

// ===========================================================================
// Fragment loaders (contiguous k order: lane half h holds k = 16h..16h+15, A and B alike)
// ===========================================================================
__device__ __forceinline__ v16h load_a_frag(const _Float16* base, int lane, int kc, int stride) {
  // A 16x32 f16: m = lane&15; lanes<16 hold K 0..15, lanes>=16 hold K 16..31
  int m  = lane & 15;
  int kb = (lane & 16) ? 16 : 0;
  const v4u* p = (const v4u*)(base + m * stride + kc * 32 + kb);  // 16B-aligned
  v8u u;
  v4u lo = p[0];   // halves kb .. kb+7
  v4u hi = p[1];   // halves kb+8 .. kb+15
#pragma unroll
  for (int i = 0; i < 4; ++i) { u[i] = lo[i]; u[4 + i] = hi[i]; }
  return __builtin_bit_cast(v16h, u);
}

__device__ __forceinline__ v16h load_b_frag(const unsigned* __restrict__ Wp, int lane,
                                            int kc, int nTiles, int nt) {
  // fragment-order layout: lane's 8 dwords are contiguous -> 2x global_load_b128
  const v8u* p = (const v8u*)(Wp + (((kc * nTiles + nt) * 32) + lane) * 8);
  return __builtin_bit_cast(v16h, *p);
}

// Dense layer: [16,K](f16,LDS) x [K,Npad] -> relu(+bias) -> [16,Npad] f16 LDS.
// A fragments hoisted into registers across all N-tiles.
template <int KC, int NT>
__device__ __forceinline__ void gemm_layer(const _Float16* in, int inStride,
                                           const unsigned* __restrict__ Wp,
                                           const float* __restrict__ bias,
                                           _Float16* out, int outStride, int lane) {
  v16h a[KC];
#pragma unroll
  for (int kc = 0; kc < KC; ++kc) a[kc] = load_a_frag(in, lane, kc, inStride);
  for (int nt = 0; nt < NT; ++nt) {
    v8f acc = {0.f, 0.f, 0.f, 0.f, 0.f, 0.f, 0.f, 0.f};
#pragma unroll
    for (int kc = 0; kc < KC; ++kc) {
      v16h b = load_b_frag(Wp, lane, kc, NT, nt);
      acc = __builtin_amdgcn_wmma_f32_16x16x32_f16(false, a[kc], false, b, (short)0, acc, false, false);
    }
    int n = nt * 16 + (lane & 15);
    float bv = bias[n];
    int mh = (lane & 16) ? 8 : 0;
#pragma unroll
    for (int r = 0; r < 8; ++r) {
      float v = acc[r] + bv;
      v = v > 0.f ? v : 0.f;
      out[(mh + r) * outStride + n] = (_Float16)v;
    }
  }
}

// Final layer: [16,K] x [K,16] -> fp32 [16][16] staging (bias for n<Nout)
template <int KC>
__device__ __forceinline__ void gemm_final(const _Float16* in, int inStride,
                                           const unsigned* __restrict__ Wp,
                                           const float* __restrict__ bias, int Nout,
                                           float* out16, int lane) {
  v8f acc = {0.f, 0.f, 0.f, 0.f, 0.f, 0.f, 0.f, 0.f};
#pragma unroll
  for (int kc = 0; kc < KC; ++kc) {
    v16h a = load_a_frag(in, lane, kc, inStride);
    v16h b = load_b_frag(Wp, lane, kc, 1, 0);
    acc = __builtin_amdgcn_wmma_f32_16x16x32_f16(false, a, false, b, (short)0, acc, false, false);
  }
  int n  = lane & 15;
  float bv = (n < Nout) ? bias[n] : 0.f;
  int mh = (lane & 16) ? 8 : 0;
#pragma unroll
  for (int r = 0; r < 8; ++r) out16[(mh + r) * 16 + n] = acc[r] + bv;
}

// ===========================================================================
// Parameters
// ===========================================================================
struct GParams {
  const float *f, *locs0, *shot0, *hit0;
  const float *by1, *by2, *by3, *bh1, *bh2, *bh3, *bs1, *bs2, *bs3;
  const float *gatW, *gata;
  const float *t0c1b, *t0c1w, *t0c2b, *t0c2w, *t0db, *t0dw;
  const float *t1c1b, *t1c1w, *t1c2b, *t1c2w;
  const float *t2c1b, *t2c1w, *t2c2b, *t2c2w, *t2db, *t2dw;
  const unsigned *Wy1, *Wy2, *Wy3, *Wh1, *Wh2, *Wh3, *Ws1, *Ws2, *Ws3;
  float *outY, *outS, *outP;
};

// ===========================================================================
// Main rollout kernel: one wave per 16 rows, all 12 steps local.
// ===========================================================================
__launch_bounds__(WAVES * 32)
__global__ void rollout_kernel(GParams P) {
  __shared__ _Float16 s_act[WAVES][16][128];   // [dec(84)|shot(16)|hit(4)|pad]
  __shared__ _Float16 s_h1 [WAVES][16][256];
  __shared__ _Float16 s_h2 [WAVES][16][128];
  __shared__ float    s_f32[WAVES][16][16];    // final-layer staging
  __shared__ float    s_z  [WAVES][16][32];    // GAT z
  __shared__ float    s_e  [WAVES][16][2];     // e_src, e_dst
  __shared__ float    s_ta [WAVES][4][32][4];  // TCN ping
  __shared__ float    s_tb [WAVES][4][32][4];  // TCN pong
  __shared__ float    s_st [WAVES][16][24];    // state: locs(4) shot(16) hit(4)

  const int tid = (int)threadIdx.x;
  const int w   = tid >> 5;
  const int l   = tid & 31;
  const int tileBase = ((int)blockIdx.x * WAVES + w) * 16;

  _Float16 (*act)[128] = s_act[w];
  _Float16 (*h1)[256]  = s_h1[w];
  _Float16 (*h2)[128]  = s_h2[w];
  float (*fsc)[16]     = s_f32[w];
  float (*zb)[32]      = s_z[w];
  float (*eb)[2]       = s_e[w];
  float (*ta)[32][4]   = s_ta[w];
  float (*tb)[32][4]   = s_tb[w];
  float (*st)[24]      = s_st[w];

  // ---- init: static features + initial state (repeat(x, 20, axis=0)) ----
  if (l < 16) {
    int row = tileBase + l;
    int src = row / 20;
#pragma unroll
    for (int c = 0; c < 16; ++c) act[l][c] = (_Float16)P.f[row * 16 + c];
#pragma unroll
    for (int c = 0; c < 4; ++c)  st[l][c]      = P.locs0[src * 4 + c];
#pragma unroll
    for (int c = 0; c < 16; ++c) st[l][4 + c]  = P.shot0[src * 16 + c];
#pragma unroll
    for (int c = 0; c < 4; ++c)  st[l][20 + c] = P.hit0[src * 4 + c];
#pragma unroll
    for (int c = 104; c < 128; ++c) act[l][c] = (_Float16)0.f;
  }
  __syncthreads();

  float tmp[32];

  for (int step = 0; step < FUTURE; ++step) {
    // ---- refresh state-dependent activation columns ----
    if (l < 16) {
#pragma unroll
      for (int c = 0; c < 4; ++c)  act[l][16 + c]  = (_Float16)st[l][c];
#pragma unroll
      for (int c = 0; c < 16; ++c) act[l][84 + c]  = (_Float16)st[l][4 + c];
#pragma unroll
      for (int c = 0; c < 4; ++c)  act[l][100 + c] = (_Float16)st[l][20 + c];
    }
    __syncthreads();

    // ================= GAT =================
    if (l < 16) {
      float z[32];
#pragma unroll
      for (int c = 0; c < 32; ++c) z[c] = 0.f;
      for (int k = 0; k < 20; ++k) {
        float x = (float)act[l][k];
#pragma unroll
        for (int c = 0; c < 32; ++c) z[c] += x * P.gatW[k * 32 + c];
      }
      float es = 0.f, ed = 0.f;
#pragma unroll
      for (int c = 0; c < 32; ++c) { es += z[c] * P.gata[c]; ed += z[c] * P.gata[32 + c]; zb[l][c] = z[c]; }
      eb[l][0] = es; eb[l][1] = ed;
    }
    __syncthreads();
    if (l < 16) {
      int g = l >> 2, a = l & 3;
      float ed = eb[l][1];
      float ev[4]; float m = -1e30f;
#pragma unroll
      for (int s = 0; s < 4; ++s) {
        float e;
        if (s == a) e = -1e9f;
        else { float t = eb[g * 4 + s][0] + ed; e = t > 0.f ? t : 0.2f * t; }
        ev[s] = e; if (e > m) m = e;
      }
      float sum = 0.f;
#pragma unroll
      for (int s = 0; s < 4; ++s) { ev[s] = __expf(ev[s] - m); sum += ev[s]; }
      float inv = 1.f / sum;
#pragma unroll
      for (int c = 0; c < 32; ++c) {
        float acc = 0.f;
#pragma unroll
        for (int s = 0; s < 4; ++s) acc += ev[s] * inv * zb[g * 4 + s][c];
        act[l][20 + c] = (_Float16)acc;
      }
    }
    __syncthreads();

    // ================= TCN (length 4 over agents) =================
    // block0 conv1: dil=1, 20 -> 16
    if (l < 16) {
      int g = l >> 2, t = l & 3;
      float o[16];
#pragma unroll
      for (int c = 0; c < 16; ++c) o[c] = P.t0c1b[c];
      for (int ci = 0; ci < 20; ++ci) {
        float xc = (float)act[g * 4 + t][ci];
        float xp = (t > 0) ? (float)act[g * 4 + t - 1][ci] : 0.f;
#pragma unroll
        for (int c = 0; c < 16; ++c)
          o[c] += P.t0c1w[(c * 20 + ci) * 2] * xp + P.t0c1w[(c * 20 + ci) * 2 + 1] * xc;
      }
#pragma unroll
      for (int c = 0; c < 16; ++c) ta[g][c][t] = o[c] > 0.f ? o[c] : 0.f;
    }
    __syncthreads();
    // block0 conv2 + down(20->16,k=1) residual  -> x1 in tb
    if (l < 16) {
      int g = l >> 2, t = l & 3;
      float o[16], r[16];
#pragma unroll
      for (int c = 0; c < 16; ++c) { o[c] = P.t0c2b[c]; r[c] = P.t0db[c]; }
      for (int ci = 0; ci < 16; ++ci) {
        float xc = ta[g][ci][t];
        float xp = (t > 0) ? ta[g][ci][t - 1] : 0.f;
#pragma unroll
        for (int c = 0; c < 16; ++c)
          o[c] += P.t0c2w[(c * 16 + ci) * 2] * xp + P.t0c2w[(c * 16 + ci) * 2 + 1] * xc;
      }
      for (int ci = 0; ci < 20; ++ci) {
        float xc = (float)act[g * 4 + t][ci];
#pragma unroll
        for (int c = 0; c < 16; ++c) r[c] += P.t0dw[c * 20 + ci] * xc;
      }
#pragma unroll
      for (int c = 0; c < 16; ++c) {
        float v = (o[c] > 0.f ? o[c] : 0.f) + r[c];
        tb[g][c][t] = v > 0.f ? v : 0.f;
      }
    }
    __syncthreads();
    // block1 conv1: dil=2, 16 -> 16  (from x1 in tb) -> ta
    if (l < 16) {
      int g = l >> 2, t = l & 3;
      float o[16];
#pragma unroll
      for (int c = 0; c < 16; ++c) o[c] = P.t1c1b[c];
      for (int ci = 0; ci < 16; ++ci) {
        float xc = tb[g][ci][t];
        float xp = (t >= 2) ? tb[g][ci][t - 2] : 0.f;
#pragma unroll
        for (int c = 0; c < 16; ++c)
          o[c] += P.t1c1w[(c * 16 + ci) * 2] * xp + P.t1c1w[(c * 16 + ci) * 2 + 1] * xc;
      }
#pragma unroll
      for (int c = 0; c < 16; ++c) ta[g][c][t] = o[c] > 0.f ? o[c] : 0.f;
    }
    __syncthreads();
    // block1 conv2 + residual x1 -> x2 (regs then tb)
    if (l < 16) {
      int g = l >> 2, t = l & 3;
      float o[16];
#pragma unroll
      for (int c = 0; c < 16; ++c) o[c] = P.t1c2b[c];
      for (int ci = 0; ci < 16; ++ci) {
        float xc = ta[g][ci][t];
        float xp = (t >= 2) ? ta[g][ci][t - 2] : 0.f;
#pragma unroll
        for (int c = 0; c < 16; ++c)
          o[c] += P.t1c2w[(c * 16 + ci) * 2] * xp + P.t1c2w[(c * 16 + ci) * 2 + 1] * xc;
      }
#pragma unroll
      for (int c = 0; c < 16; ++c) {
        float v = (o[c] > 0.f ? o[c] : 0.f) + tb[g][c][t];
        tmp[c] = v > 0.f ? v : 0.f;
      }
    }
    __syncthreads();
    if (l < 16) {
      int g = l >> 2, t = l & 3;
#pragma unroll
      for (int c = 0; c < 16; ++c) tb[g][c][t] = tmp[c];   // x2 -> tb
    }
    __syncthreads();
    // block2 conv1: dil=4 (t-4 < 0 => only k=1 tap), 16 -> 32 -> ta
    if (l < 16) {
      int g = l >> 2, t = l & 3;
      float o[32];
#pragma unroll
      for (int c = 0; c < 32; ++c) o[c] = P.t2c1b[c];
      for (int ci = 0; ci < 16; ++ci) {
        float xc = tb[g][ci][t];
#pragma unroll
        for (int c = 0; c < 32; ++c) o[c] += P.t2c1w[(c * 16 + ci) * 2 + 1] * xc;
      }
#pragma unroll
      for (int c = 0; c < 32; ++c) ta[g][c][t] = o[c] > 0.f ? o[c] : 0.f;
    }
    __syncthreads();
    // block2 conv2 (dil=4, 32->32) + down(16->32) -> x3 (regs then ta)
    if (l < 16) {
      int g = l >> 2, t = l & 3;
      float o[32], r[32];
#pragma unroll
      for (int c = 0; c < 32; ++c) { o[c] = P.t2c2b[c]; r[c] = P.t2db[c]; }
      for (int ci = 0; ci < 32; ++ci) {
        float xc = ta[g][ci][t];
#pragma unroll
        for (int c = 0; c < 32; ++c) o[c] += P.t2c2w[(c * 32 + ci) * 2 + 1] * xc;
      }
      for (int ci = 0; ci < 16; ++ci) {
        float xc = tb[g][ci][t];
#pragma unroll
        for (int c = 0; c < 32; ++c) r[c] += P.t2dw[c * 16 + ci] * xc;
      }
#pragma unroll
      for (int c = 0; c < 32; ++c) {
        float v = (o[c] > 0.f ? o[c] : 0.f) + r[c];
        tmp[c] = v > 0.f ? v : 0.f;
      }
    }
    __syncthreads();
    if (l < 16) {
      int g = l >> 2, t = l & 3;
#pragma unroll
      for (int c = 0; c < 32; ++c) ta[g][c][t] = tmp[c];   // x3 -> ta
    }
    __syncthreads();
    // f_cov: [B,32,4].reshape(N,32): row r gets (c=8r+j, t') flattened
    if (l < 16) {
      int g = l >> 2, r = l & 3;
#pragma unroll
      for (int j = 0; j < 8; ++j)
#pragma unroll
        for (int t2 = 0; t2 < 4; ++t2)
          act[l][52 + j * 4 + t2] = (_Float16)ta[g][8 * r + j][t2];
    }
    __syncthreads();

    // ================= decoder dec_y (WMMA) =================
    gemm_layer<4, 16>(&act[0][0], 128, P.Wy1, P.by1, &h1[0][0], 256, l);
    gemm_layer<8, 8>(&h1[0][0], 256, P.Wy2, P.by2, &h2[0][0], 128, l);
    gemm_final<4>(&h2[0][0], 128, P.Wy3, P.by3, 4, &fsc[0][0], l);
    __syncthreads();
    if (l < 16) {
      int row = tileBase + l;
      float y[4];
#pragma unroll
      for (int n = 0; n < 4; ++n) y[n] = fsc[l][n] + st[l][n];
      P.outY[(row * 12 + step) * 2 + 0] = y[0];
      P.outY[(row * 12 + step) * 2 + 1] = y[1];
#pragma unroll
      for (int n = 0; n < 4; ++n) st[l][n] = y[n];
    }
    __syncthreads();

    // ================= decoder dec_hit (WMMA + softmax4) =================
    gemm_layer<4, 16>(&act[0][0], 128, P.Wh1, P.bh1, &h1[0][0], 256, l);
    gemm_layer<8, 8>(&h1[0][0], 256, P.Wh2, P.bh2, &h2[0][0], 128, l);
    gemm_final<4>(&h2[0][0], 128, P.Wh3, P.bh3, 4, &fsc[0][0], l);
    __syncthreads();
    if (l < 16) {
      int row = tileBase + l;
      float e[4], m = -1e30f, sum = 0.f;
#pragma unroll
      for (int n = 0; n < 4; ++n) { e[n] = fsc[l][n]; if (e[n] > m) m = e[n]; }
#pragma unroll
      for (int n = 0; n < 4; ++n) { e[n] = __expf(e[n] - m); sum += e[n]; }
      float inv = 1.f / sum;
#pragma unroll
      for (int n = 0; n < 4; ++n) {
        float p = e[n] * inv;
        P.outP[(row * 12 + step) * 4 + n] = p;
        st[l][20 + n] = p;
      }
    }
    __syncthreads();

    // ================= decoder dec_shot (WMMA + softmax16) =================
    gemm_layer<4, 16>(&act[0][0], 128, P.Ws1, P.bs1, &h1[0][0], 256, l);
    gemm_layer<8, 8>(&h1[0][0], 256, P.Ws2, P.bs2, &h2[0][0], 128, l);
    gemm_final<4>(&h2[0][0], 128, P.Ws3, P.bs3, 16, &fsc[0][0], l);
    __syncthreads();
    if (l < 16) {
      int row = tileBase + l;
      float e[16], m = -1e30f, sum = 0.f;
#pragma unroll
      for (int n = 0; n < 16; ++n) { e[n] = fsc[l][n]; if (e[n] > m) m = e[n]; }
#pragma unroll
      for (int n = 0; n < 16; ++n) { e[n] = __expf(e[n] - m); sum += e[n]; }
      float inv = 1.f / sum;
#pragma unroll
      for (int n = 0; n < 16; ++n) {
        float s = e[n] * inv;
        P.outS[(row * 12 + step) * 16 + n] = s;
        st[l][4 + n] = s;
      }
    }
    __syncthreads();
  }
}

// ===========================================================================
// Launch
// ===========================================================================
extern "C" void kernel_launch(void* const* d_in, const int* in_sizes, int n_in,
                              void* d_out, int out_size, void* d_ws, size_t ws_size,
                              hipStream_t stream) {
  (void)in_sizes; (void)n_in; (void)out_size; (void)ws_size;
  const float* F = (const float*)d_in[0];
  unsigned* ws = (unsigned*)d_ws;

  // ---- pack fp32 weights -> fragment-order f16 pair layout ----
  struct PackJob { int src; unsigned off; int K, N, Kpad, Npad, remap; };
  // Leaf order (sorted dict keys): dec_hit, dec_shot, dec_y, gat_W, gat_a, tcn
  const PackJob jobs[9] = {
    {  4, OFF_H1,  88, 256, 128, 256, 1 },   // dec_hit L0 (rows 84..87 -> 100..103)
    {  6, OFF_H2, 256, 128, 256, 128, 0 },
    {  8, OFF_H3, 128,   4, 128,  16, 0 },
    { 10, OFF_S1, 104, 256, 128, 256, 0 },   // dec_shot L0
    { 12, OFF_S2, 256, 128, 256, 128, 0 },
    { 14, OFF_S3, 128,  16, 128,  16, 0 },
    { 16, OFF_Y1,  84, 256, 128, 256, 0 },   // dec_y L0
    { 18, OFF_Y2, 256, 128, 256, 128, 0 },
    { 20, OFF_Y3, 128,   4, 128,  16, 0 },
  };
  for (int j = 0; j < 9; ++j) {
    int total = (jobs[j].Kpad / 2) * jobs[j].Npad;
    pack_w_kernel<<<(total + 255) / 256, 256, 0, stream>>>(
        (const float*)d_in[jobs[j].src], ws + jobs[j].off,
        jobs[j].K, jobs[j].N, jobs[j].Kpad, jobs[j].Npad, jobs[j].remap);
  }

  // ---- main rollout ----
  GParams P;
  P.f = F; P.locs0 = (const float*)d_in[1];
  P.shot0 = (const float*)d_in[2]; P.hit0 = (const float*)d_in[3];
  P.bh1 = (const float*)d_in[5];  P.bh2 = (const float*)d_in[7];  P.bh3 = (const float*)d_in[9];
  P.bs1 = (const float*)d_in[11]; P.bs2 = (const float*)d_in[13]; P.bs3 = (const float*)d_in[15];
  P.by1 = (const float*)d_in[17]; P.by2 = (const float*)d_in[19]; P.by3 = (const float*)d_in[21];
  P.gatW = (const float*)d_in[22]; P.gata = (const float*)d_in[23];
  P.t0c1b = (const float*)d_in[24]; P.t0c1w = (const float*)d_in[25];
  P.t0c2b = (const float*)d_in[26]; P.t0c2w = (const float*)d_in[27];
  P.t0db  = (const float*)d_in[28]; P.t0dw  = (const float*)d_in[29];
  P.t1c1b = (const float*)d_in[30]; P.t1c1w = (const float*)d_in[31];
  P.t1c2b = (const float*)d_in[32]; P.t1c2w = (const float*)d_in[33];
  P.t2c1b = (const float*)d_in[34]; P.t2c1w = (const float*)d_in[35];
  P.t2c2b = (const float*)d_in[36]; P.t2c2w = (const float*)d_in[37];
  P.t2db  = (const float*)d_in[38]; P.t2dw  = (const float*)d_in[39];
  P.Wy1 = ws + OFF_Y1; P.Wy2 = ws + OFF_Y2; P.Wy3 = ws + OFF_Y3;
  P.Wh1 = ws + OFF_H1; P.Wh2 = ws + OFF_H2; P.Wh3 = ws + OFF_H3;
  P.Ws1 = ws + OFF_S1; P.Ws2 = ws + OFF_S2; P.Ws3 = ws + OFF_S3;
  float* out = (float*)d_out;
  P.outY = out;                                   // [N,12,2]
  P.outS = out + (size_t)N_ROWS * 12 * 2;         // [N,12,16]
  P.outP = out + (size_t)N_ROWS * 12 * (2 + 16);  // [N,12,4]

  const int rowsPerBlock = WAVES * 16;
  rollout_kernel<<<N_ROWS / rowsPerBlock, WAVES * 32, 0, stream>>>(P);
}
